// CodeformerLM_4466765988404
// MI455X (gfx1250) — matrix operand order repaired and checked by reading, not compiled, not run on hardware
//
#include <hip/hip_runtime.h>
#include <hip/hip_bf16.h>
#include <cstdint>
#include <cstddef>

// ---------------------------------------------------------------------------
// Effective computation (chunk_units / sos are dead code w.r.t. the output):
//   u[n,:]      = mask(n) ? E[token[n]] @ W_dec : 0        n in [0, 2048)
//   logits[n,:] = u[n,:] @ E^T                              (2048 x 32000)
// GEMM2 (100.7 GFLOP) dominates; bf16 WMMA + TDM-fed LDS double buffering.
// ---------------------------------------------------------------------------

typedef __attribute__((ext_vector_type(16))) __bf16        v16bf;
typedef __attribute__((ext_vector_type(8)))  float         v8f;
typedef __attribute__((ext_vector_type(4)))  unsigned int  u32x4;
typedef __attribute__((ext_vector_type(8)))  int           i32x8;
typedef __attribute__((ext_vector_type(4)))  int           i32x4;

#if defined(__has_builtin)
#if __has_builtin(__builtin_amdgcn_tensor_load_to_lds) && \
    __has_builtin(__builtin_amdgcn_s_wait_tensorcnt)
#define HAS_TDM 1
#else
#define HAS_TDM 0
#endif
#else
#define HAS_TDM 0
#endif

namespace cfg {
constexpr int Bb  = 4;
constexpr int Cc  = 16;
constexpr int Tt  = 33;
constexpr int Hh  = 768;
constexpr int Vv  = 32000;
constexpr int TM1 = Tt - 1;              // 32
constexpr int NR  = Bb * Cc * TM1;       // 2048 output rows
constexpr int BM  = 128;                 // block tile M
constexpr int BN  = 128;                 // block tile N
constexpr int BK  = 64;                  // block tile K (2 wmma K-steps)
constexpr int BKP = BK + 8;              // LDS row stride in halfs (144B)
constexpr int TILE_HALFS = BM * BKP;     // one padded tile (A or B)
}

__device__ __forceinline__ unsigned short f2bf(float f) {
  // round-to-nearest-even fp32 -> bf16
  unsigned u = __builtin_bit_cast(unsigned, f);
  u += 0x7fffu + ((u >> 16) & 1u);
  return (unsigned short)(u >> 16);
}

union Frag {
  uint4 q[2];
  v16bf v;
};

// ---------------------------------------------------------------------------
// TDM: async 2D tile DMA  global -> LDS  (CDNA5 ISA ch.8, D# groups 0..3).
// Tile: tile_d1 rows x tile_d0 bf16 elements, row stride `row_stride` elems.
// LDS padding: after every 32 DWORDs (=64 halfs = one tile row) insert
// 4 DWORDs (=8 halfs) -> LDS row pitch = BKP = 72 halfs, matching the
// bank-conflict-free fragment layout used by the compute waves.
// ---------------------------------------------------------------------------
#if HAS_TDM
__device__ __forceinline__ void tdm_load_2d(const void* gptr, unsigned lds_byte,
                                            int tensor_d0, int tensor_d1,
                                            int tile_d0, int tile_d1,
                                            int row_stride) {
  unsigned long long ga = (unsigned long long)gptr;
  u32x4 g0;
  g0[0] = 1u;                                         // count=1, user descriptor
  g0[1] = lds_byte;                                   // lds_addr
  g0[2] = (unsigned)(ga & 0xffffffffu);               // global_addr[31:0]
  g0[3] = (unsigned)((ga >> 32) & 0x01ffffffu)        // global_addr[56:32]
        | (2u << 30);                                 // type = 2 ("image")
  i32x8 g1;
  g1[0] = (1 << 16)                                   // data_size = 2 bytes
        | (1 << 20)                                   // pad_enable
        | (4 << 22)                                   // pad_interval = 32 dwords
        | (3 << 25);                                  // pad_amount  = 4 dwords
  g1[1] = (tensor_d0 & 0xffff) << 16;                 // tensor_dim0[15:0]
  g1[2] = ((tensor_d0 >> 16) & 0xffff)                // tensor_dim0[31:16]
        | ((tensor_d1 & 0xffff) << 16);               // tensor_dim1[15:0]
  g1[3] = ((tensor_d1 >> 16) & 0xffff)                // tensor_dim1[31:16]
        | ((tile_d0 & 0xffff) << 16);                 // tile_dim0
  g1[4] = (tile_d1 & 0xffff);                         // tile_dim1 (tile_dim2=0)
  g1[5] = row_stride;                                 // tensor_dim0_stride[31:0]
  g1[6] = 0;
  g1[7] = 0;
  const i32x4 gz = {0, 0, 0, 0};
#if __clang_major__ >= 23
  const i32x8 gz8 = {0, 0, 0, 0, 0, 0, 0, 0};
  __builtin_amdgcn_tensor_load_to_lds(g0, g1, gz, gz, gz8, 0);
#else
  __builtin_amdgcn_tensor_load_to_lds(g0, g1, gz, gz, 0);
#endif
}
#endif

// ---------------------------------------------------------------------------
// fp32 -> bf16 bulk convert (vectorized by 4)
// ---------------------------------------------------------------------------
__global__ __launch_bounds__(256)
void cvt_f32_to_bf16(const float* __restrict__ src,
                     unsigned short* __restrict__ dst, int n4) {
  int i = blockIdx.x * 256 + threadIdx.x;
  if (i >= n4) return;
  float4 f = ((const float4*)src)[i];
  ushort4 o;
  o.x = f2bf(f.x); o.y = f2bf(f.y); o.z = f2bf(f.z); o.w = f2bf(f.w);
  ((ushort4*)dst)[i] = o;
}

// ---------------------------------------------------------------------------
// Gather token embedding rows, apply validity mask, convert to bf16.
// ---------------------------------------------------------------------------
__global__ __launch_bounds__(192)
void gather_tokens_bf16(const int* __restrict__ tok_ids,
                        const int* __restrict__ split_sizes,
                        const int* __restrict__ num_chunks,
                        const float* __restrict__ E,
                        unsigned short* __restrict__ A1) {
  using namespace cfg;
  const int n   = blockIdx.x;
  const int b   = n >> 9;
  const int rem = n & 511;
  const int c   = rem >> 5;
  const int t   = rem & 31;
  const bool valid = (t < split_sizes[b * Cc + c]) && (c < num_chunks[b]);
  const int  tok   = tok_ids[(b * Cc + c) * Tt + t];

  ushort4* dst = (ushort4*)(A1 + (size_t)n * Hh);
  const int i = threadIdx.x;                      // 192 threads x float4 = 768
  ushort4 o;
  if (valid) {
    float4 f = ((const float4*)(E + (size_t)tok * Hh))[i];
    o.x = f2bf(f.x); o.y = f2bf(f.y); o.z = f2bf(f.z); o.w = f2bf(f.w);
  } else {
    o.x = 0; o.y = 0; o.z = 0; o.w = 0;
  }
  dst[i] = o;
}

// ---------------------------------------------------------------------------
// GEMM1 (2.4% of FLOPs): C[M,N](bf16) = A[M,K](bf16) * B[K,N](bf16 row-major)
// Single-buffer LDS staging, 8 waves of 64x32, v_wmma_f32_16x16x32_bf16.
// ---------------------------------------------------------------------------
__global__ __launch_bounds__(256)
void gemm1_bf16_wmma(const unsigned short* __restrict__ A,
                     const unsigned short* __restrict__ B,
                     unsigned short* __restrict__ Cout,
                     int M, int N, int K) {
  using namespace cfg;
  __shared__ unsigned short As[BM][BKP];
  __shared__ unsigned short Bs[BN][BKP];

  const int tid   = threadIdx.x;
  const int lane  = tid & 31;
  const int wave  = tid >> 5;
  const int wm    = wave >> 2;
  const int wn    = wave & 3;
  const int m0    = blockIdx.y * BM;
  const int n0    = blockIdx.x * BN;
  const int lhalf = (lane < 16) ? 0 : 1;
  const int l15   = lane & 15;

  v8f acc[4][2];
  const v8f vzero = {0.f, 0.f, 0.f, 0.f, 0.f, 0.f, 0.f, 0.f};
#pragma unroll
  for (int i = 0; i < 4; ++i)
#pragma unroll
    for (int j = 0; j < 2; ++j) acc[i][j] = vzero;

  for (int k0 = 0; k0 < K; k0 += BK) {
#pragma unroll
    for (int i = 0; i < 4; ++i) {
      int chunk = tid + i * 256;
      int r  = chunk >> 3;
      int cc = (chunk & 7) << 3;
      uint4 val = *(const uint4*)(A + (size_t)(m0 + r) * K + (k0 + cc));
      *(uint4*)&As[r][cc] = val;
    }
#pragma unroll
    for (int i = 0; i < 4; ++i) {                  // transpose K-major B into LDS
      int chunk = tid + i * 256;
      int kk = chunk >> 4;
      int nn = (chunk & 15) << 3;
      uint4 val = *(const uint4*)(B + (size_t)(k0 + kk) * N + (n0 + nn));
      const unsigned short* p = (const unsigned short*)&val;
#pragma unroll
      for (int j = 0; j < 8; ++j) Bs[nn + j][kk] = p[j];
    }
    __syncthreads();

#pragma unroll
    for (int kk = 0; kk < BK / 32; ++kk) {
      Frag fa[4], fb[2];
#pragma unroll
      for (int fm = 0; fm < 4; ++fm) {
        int row = wm * 64 + fm * 16 + l15;
        int kh  = kk * 32 + lhalf * 8;
        fa[fm].q[0] = *(const uint4*)&As[row][kh];
        fa[fm].q[1] = *(const uint4*)&As[row][kh + 16];
      }
#pragma unroll
      for (int fn = 0; fn < 2; ++fn) {
        int col = wn * 32 + fn * 16 + l15;
        int kb  = kk * 32 + lhalf * 8;
        fb[fn].q[0] = *(const uint4*)&Bs[col][kb];
        fb[fn].q[1] = *(const uint4*)&Bs[col][kb + 16];
      }
#pragma unroll
      for (int fm = 0; fm < 4; ++fm)
#pragma unroll
        for (int fn = 0; fn < 2; ++fn)
          acc[fm][fn] = __builtin_amdgcn_wmma_f32_16x16x32_bf16(
              false, fa[fm].v, false, fb[fn].v, (short)0, acc[fm][fn],
              false, false);
    }
    __syncthreads();
  }

#pragma unroll
  for (int fm = 0; fm < 4; ++fm)
#pragma unroll
    for (int fn = 0; fn < 2; ++fn) {
      int row = m0 + wm * 64 + fm * 16 + lhalf * 8;
      int col = n0 + wn * 32 + fn * 16 + l15;
#pragma unroll
      for (int r = 0; r < 8; ++r)
        Cout[(size_t)(row + r) * N + col] = f2bf(acc[fm][fn][r]);
    }
}

// ---------------------------------------------------------------------------
// GEMM2 (dominant): C[M,N](f32) = A[M,K](bf16) * B[N,K](bf16 row-major)^T
// TDM-fed, double-buffered LDS pipeline:
//   wave0: s_wait_tensorcnt 0  -> barrier publishes stage kt
//   wave0: issue TDM for stage kt+1 into the other buffer (overlaps compute;
//          safe: all readers of that buffer finished before this barrier)
//   all waves: 16x v_wmma on stage kt, no second barrier needed.
// Dynamic LDS (73.7 KB, legal on the 320 KB CDNA5 WGP).
// ---------------------------------------------------------------------------
__global__ __launch_bounds__(256)
void gemm2_tdm_wmma(const unsigned short* __restrict__ A,
                    const unsigned short* __restrict__ Bm,
                    float* __restrict__ Cout,
                    int M, int N, int K) {
  using namespace cfg;
  extern __shared__ __align__(16) unsigned short smem[];
  unsigned short* As0 = smem;                        // [2][BM][BKP]
  unsigned short* Bs0 = smem + 2 * TILE_HALFS;       // [2][BN][BKP]

  const int tid   = threadIdx.x;
  const int lane  = tid & 31;
  const int wave  = tid >> 5;
  const int wm    = wave >> 2;
  const int wn    = wave & 3;
  const int m0    = blockIdx.y * BM;
  const int n0    = blockIdx.x * BN;
  const int lhalf = (lane < 16) ? 0 : 1;
  const int l15   = lane & 15;
  const int nK    = K / BK;

  v8f acc[4][2];
  const v8f vzero = {0.f, 0.f, 0.f, 0.f, 0.f, 0.f, 0.f, 0.f};
#pragma unroll
  for (int i = 0; i < 4; ++i)
#pragma unroll
    for (int j = 0; j < 2; ++j) acc[i][j] = vzero;

#if HAS_TDM
  auto issue_stage = [&](int kt, int buf) {
    tdm_load_2d(A  + (size_t)m0 * K + kt * BK,
                (unsigned)(unsigned long long)(As0 + buf * TILE_HALFS),
                K, M, BK, BM, K);
    tdm_load_2d(Bm + (size_t)n0 * K + kt * BK,
                (unsigned)(unsigned long long)(Bs0 + buf * TILE_HALFS),
                K, N, BK, BN, K);
  };
  if (wave == 0) issue_stage(0, 0);
#endif

  for (int kt = 0; kt < nK; ++kt) {
    const int buf = kt & 1;
    const unsigned short* As_ = As0 + buf * TILE_HALFS;
    const unsigned short* Bs_ = Bs0 + buf * TILE_HALFS;

#if HAS_TDM
    if (wave == 0) __builtin_amdgcn_s_wait_tensorcnt(0);
    __syncthreads();                               // publish stage kt to all waves
    if (wave == 0 && kt + 1 < nK) issue_stage(kt + 1, buf ^ 1);
#else
    // fallback: manual cooperative staging (round-1 style)
    {
      unsigned short* Aw = As0 + buf * TILE_HALFS;
      unsigned short* Bw = Bs0 + buf * TILE_HALFS;
#pragma unroll
      for (int i = 0; i < 4; ++i) {
        int chunk = tid + i * 256;
        int r  = chunk >> 3;
        int cc = (chunk & 7) << 3;
        uint4 va = *(const uint4*)(A  + (size_t)(m0 + r) * K + (kt * BK + cc));
        uint4 vb = *(const uint4*)(Bm + (size_t)(n0 + r) * K + (kt * BK + cc));
        *(uint4*)&Aw[r * BKP + cc] = va;
        *(uint4*)&Bw[r * BKP + cc] = vb;
      }
    }
    __syncthreads();
#endif

#pragma unroll
    for (int kk = 0; kk < BK / 32; ++kk) {
      Frag fa[4], fb[2];
#pragma unroll
      for (int fm = 0; fm < 4; ++fm) {
        int row = wm * 64 + fm * 16 + l15;
        int kh  = kk * 32 + lhalf * 8;
        fa[fm].q[0] = *(const uint4*)&As_[row * BKP + kh];
        fa[fm].q[1] = *(const uint4*)&As_[row * BKP + kh + 16];
      }
#pragma unroll
      for (int fn = 0; fn < 2; ++fn) {
        int col = wn * 32 + fn * 16 + l15;
        int kb  = kk * 32 + lhalf * 8;
        fb[fn].q[0] = *(const uint4*)&Bs_[col * BKP + kb];
        fb[fn].q[1] = *(const uint4*)&Bs_[col * BKP + kb + 16];
      }
#pragma unroll
      for (int fm = 0; fm < 4; ++fm)
#pragma unroll
        for (int fn = 0; fn < 2; ++fn)
          acc[fm][fn] = __builtin_amdgcn_wmma_f32_16x16x32_bf16(
              false, fa[fm].v, false, fb[fn].v, (short)0, acc[fm][fn],
              false, false);
    }
#if !HAS_TDM
    __syncthreads();
#endif
  }

  // epilogue: 262 MB write-once stream -> non-temporal stores
#pragma unroll
  for (int fm = 0; fm < 4; ++fm)
#pragma unroll
    for (int fn = 0; fn < 2; ++fn) {
      int row = m0 + wm * 64 + fm * 16 + lhalf * 8;
      int col = n0 + wn * 32 + fn * 16 + l15;
#pragma unroll
      for (int r = 0; r < 8; ++r)
        __builtin_nontemporal_store(acc[fm][fn][r],
                                    Cout + (size_t)(row + r) * N + col);
    }
}

// ---------------------------------------------------------------------------
// Entry point
// ---------------------------------------------------------------------------
extern "C" void kernel_launch(void* const* d_in, const int* in_sizes, int n_in,
                              void* d_out, int out_size, void* d_ws, size_t ws_size,
                              hipStream_t stream) {
  using namespace cfg;
  (void)in_sizes; (void)n_in; (void)out_size; (void)ws_size;

  const int*   tok_ids  = (const int*)d_in[0];    // (B,C,T)
  const int*   splits   = (const int*)d_in[1];    // (B,C)
  const int*   nchunks  = (const int*)d_in[2];    // (B,)
  const float* E        = (const float*)d_in[4];  // (V,H)
  const float* Wdec     = (const float*)d_in[6];  // (H,H)
  float*       out      = (float*)d_out;          // (B,C,32,V)

  unsigned short* Ebf = (unsigned short*)d_ws;            // V*H bf16 (~49 MB)
  unsigned short* Wbf = Ebf + (size_t)Vv * Hh;            // H*H bf16
  unsigned short* A1  = Wbf + (size_t)Hh * Hh;            // NR*H bf16
  unsigned short* U   = A1  + (size_t)NR * Hh;            // NR*H bf16

  {
    int n4 = (Vv * Hh) / 4;
    cvt_f32_to_bf16<<<(n4 + 255) / 256, 256, 0, stream>>>(E, Ebf, n4);
  }
  {
    int n4 = (Hh * Hh) / 4;
    cvt_f32_to_bf16<<<(n4 + 255) / 256, 256, 0, stream>>>(Wdec, Wbf, n4);
  }

  gather_tokens_bf16<<<NR, 192, 0, stream>>>(tok_ids, splits, nchunks, E, A1);

  // GEMM1: U[2048,768] = A1 @ W_dec
  gemm1_bf16_wmma<<<dim3(Hh / BN, NR / BM), 256, 0, stream>>>(
      A1, Wbf, U, NR, Hh, Hh);

  // GEMM2: logits[2048,32000] = U @ E^T  (TDM double-buffered)
  const size_t smem_bytes = (size_t)4 * TILE_HALFS * sizeof(unsigned short);
  gemm2_tdm_wmma<<<dim3(Vv / BN, NR / BM), 256, smem_bytes, stream>>>(
      U, Ebf, out, NR, Vv, Hh);
}
